// RSPLayer_89464168776149
// MI455X (gfx1250) — matrix operand order repaired and checked by reading, not compiled, not run on hardware
//
#include <hip/hip_runtime.h>
#include <hip/hip_bf16.h>
#include <cstdint>

typedef __attribute__((ext_vector_type(16))) __bf16 v16bf;
typedef __attribute__((ext_vector_type(8)))  __bf16 v8bf;
typedef __attribute__((ext_vector_type(8)))  float  v8f;

#define DDIM   1024
#define MTOK   8192   // B*L
#define BQ     4
#define LSEQ   2048
#define CHUNKS 64
#define CLEN   32
#define LDT    72     // padded LDS tile stride (elements), BK=64 + 8 pad

#if defined(__HIP_DEVICE_COMPILE__) && __has_builtin(__builtin_amdgcn_global_load_async_to_lds_b128)
#define HAVE_ASYNC 1
#else
#define HAVE_ASYNC 0
#endif

#if HAVE_ASYNC
typedef int v4i_ __attribute__((vector_size(16)));
typedef v4i_ __attribute__((address_space(1))) * gv4i_p;   // global int4*
typedef v4i_ __attribute__((address_space(3))) * lv4i_p;   // lds int4*
#endif

__device__ __forceinline__ void async_copy16(const __bf16* gsrc, __bf16* ldst) {
#if HAVE_ASYNC
  __builtin_amdgcn_global_load_async_to_lds_b128(
      (gv4i_p)(uintptr_t)gsrc, (lv4i_p)(uint32_t)(uintptr_t)ldst, 0, 0);
#else
  *(uint4*)ldst = *(const uint4*)gsrc;
#endif
}

__device__ __forceinline__ void wait_async0() {
#if HAVE_ASYNC
#if __has_builtin(__builtin_amdgcn_s_wait_asynccnt)
  __builtin_amdgcn_s_wait_asynccnt(0);
#else
  asm volatile("s_wait_asynccnt 0" ::: "memory");
#endif
#endif
}

__device__ __forceinline__ float sigmoidf_(float x) {
  return 1.0f / (1.0f + __expf(-x));
}

// ---------------- WMMA fragment loaders (wave32 layouts, ISA 7.12.2) -------
// A 16x32 bf16: M = lane&15, two contiguous 8-elem (16B) chunks per lane.
__device__ __forceinline__ v16bf load_fragA(const __bf16* tile, int lane) {
  int m = lane & 15, hk = lane >> 4;
  const __bf16* p = tile + m * LDT;
  v8bf lo = *(const v8bf*)(p + 8 * hk);
  v8bf hi = *(const v8bf*)(p + 16 + 8 * hk);
  return __builtin_shufflevector(lo, hi, 0,1,2,3,4,5,6,7,8,9,10,11,12,13,14,15);
}
// B 32x16 bf16: N = lane&15, two 8-elem chunks at K = 8*(lane>=16) and 16 + 8*(lane>=16)   (sB stored [n][k])
__device__ __forceinline__ v16bf load_fragB(const __bf16* tile, int lane) {
  int n = lane & 15, hk = lane >> 4;
  const __bf16* p = tile + n * LDT;
  v8bf lo = *(const v8bf*)(p + 8 * hk);
  v8bf hi = *(const v8bf*)(p + 16 + 8 * hk);
  return __builtin_shufflevector(lo, hi, 0,1,2,3,4,5,6,7,8,9,10,11,12,13,14,15);
}

// ---------------- fp32 -> bf16 pre-convert ---------------------------------
__global__ __launch_bounds__(256) void cvt_f32_bf16(
    const float* __restrict__ s, __bf16* __restrict__ d) {
  const int i = (blockIdx.x * 256 + threadIdx.x) * 8;
  float4 a = *(const float4*)(s + i);
  float4 b = *(const float4*)(s + i + 4);
  v8bf o;
  o[0]=(__bf16)a.x; o[1]=(__bf16)a.y; o[2]=(__bf16)a.z; o[3]=(__bf16)a.w;
  o[4]=(__bf16)b.x; o[5]=(__bf16)b.y; o[6]=(__bf16)b.z; o[7]=(__bf16)b.w;
  *(v8bf*)(d + i) = o;
}

// ---------------- fused GEMM:  Out[row,col] = epi( sum_k A[row,k]*W[col,k] )
// EPI: 0 lambda, 1 phi(beta), 2 alpha, 3 rho->h (bf16 out), 4 residual out
template <int EPI, bool CONCAT>
__global__ __launch_bounds__(256) void wmma_gemm_epi(
    const __bf16* __restrict__ A,   // [M,K] bf16 when !CONCAT
    const __bf16* __restrict__ A1,  // f (K seg 0, stride DDIM) when CONCAT
    const __bf16* __restrict__ A2,  // s (seg 1)
    const __bf16* __restrict__ A3,  // x (seg 2)
    const __bf16* __restrict__ W,   // [N,K] bf16 row-major
    const float* __restrict__ bias,
    const float* __restrict__ fbuf, const float* __restrict__ sbuf,
    const float* __restrict__ xres, const float* __restrict__ rscale,
    float* __restrict__ OutF, __bf16* __restrict__ OutH, int K)
{
  __shared__ __align__(16) __bf16 sA[2][64][LDT];
  __shared__ __align__(16) __bf16 sB[2][128][LDT];

  const int tid  = threadIdx.x;
  const int lane = tid & 31, wave = tid >> 5;
  const int wm = wave >> 2, wn = wave & 3;          // 2x4 waves -> 64x128 tile
  const int bM = blockIdx.x * 64, bN = blockIdx.y * 128;

  v8f acc[2][2] = {};

  const int NT = K >> 6;   // 64-wide K tiles

  // per-thread chunk coords (8 bf16 = 16B chunks, 8 chunks per 64-elem row)
  auto stage = [&](int buf, int k0) {
    const __bf16* Ap; int ks, lda;
    if constexpr (CONCAT) {
      int seg = k0 >> 10;
      Ap = (seg == 0) ? A1 : (seg == 1) ? A2 : A3;
      ks = k0 & 1023; lda = DDIM;
    } else { Ap = A; ks = k0; lda = K; }
#pragma unroll
    for (int t = 0; t < 2; ++t) {                  // A: 512 chunks / 256 thr
      int c = tid + t * 256;
      int row = c >> 3, cc = (c & 7) * 8;
      async_copy16(Ap + (size_t)(bM + row) * lda + ks + cc, &sA[buf][row][cc]);
    }
#pragma unroll
    for (int t = 0; t < 4; ++t) {                  // W: 1024 chunks / 256 thr
      int c = tid + t * 256;
      int row = c >> 3, cc = (c & 7) * 8;
      async_copy16(W + (size_t)(bN + row) * K + k0 + cc, &sB[buf][row][cc]);
    }
  };

  stage(0, 0);
  for (int i = 0; i < NT; ++i) {
    wait_async0();
    __syncthreads();
    if (i + 1 < NT) stage((i + 1) & 1, (i + 1) << 6);

    const __bf16* tA = &sA[i & 1][0][0];
    const __bf16* tB = &sB[i & 1][0][0];
#pragma unroll
    for (int kk = 0; kk < 2; ++kk) {
      v16bf af[2], bfm[2];
#pragma unroll
      for (int mi = 0; mi < 2; ++mi)
        af[mi] = load_fragA(tA + (wm*32 + mi*16) * LDT + kk*32, lane);
#pragma unroll
      for (int ni = 0; ni < 2; ++ni)
        bfm[ni] = load_fragB(tB + (wn*32 + ni*16) * LDT + kk*32, lane);
#pragma unroll
      for (int mi = 0; mi < 2; ++mi)
#pragma unroll
        for (int ni = 0; ni < 2; ++ni)
          acc[mi][ni] = __builtin_amdgcn_wmma_f32_16x16x32_bf16(
              false, af[mi], false, bfm[ni], (short)0, acc[mi][ni], false, false);
    }
  }

  // ---- epilogue (C/D layout: VGPR r -> M = r + 8*(lane>=16), N = lane&15) --
  const int l15 = lane & 15, hi8 = (lane >> 4) * 8;
  float rs = 0.0f;
  if constexpr (EPI == 4) rs = rscale[0];
#pragma unroll
  for (int mi = 0; mi < 2; ++mi) {
#pragma unroll
    for (int ni = 0; ni < 2; ++ni) {
      const int col = bN + wn*32 + ni*16 + l15;
      float bv = 0.0f;
      if constexpr (EPI == 0 || EPI == 2 || EPI == 3) bv = bias[col];
#pragma unroll
      for (int r = 0; r < 8; ++r) {
        const int row = bM + wm*32 + mi*16 + hi8 + r;
        const size_t idx = (size_t)row * DDIM + col;
        float v = acc[mi][ni][r];
        if constexpr (EPI == 0) {            // lambda
          v = fminf(fmaxf(v + bv, -4.0f), 4.0f);
          v = __expf(-__expf(v));
          v = fminf(fmaxf(v, 0.001f), 0.999f);
          OutF[idx] = v;
        } else if constexpr (EPI == 1) {     // phi = silu + 0.1 sin
          OutF[idx] = v * sigmoidf_(v) + 0.1f * __sinf(v);
        } else if constexpr (EPI == 2) {     // alpha
          OutF[idx] = sigmoidf_(v + bv);
        } else if constexpr (EPI == 3) {     // rho -> h (bf16 for next GEMM)
          float rho = sigmoidf_(v + bv);
          OutH[idx] = (__bf16)(rho * fbuf[idx] + (1.0f - rho) * sbuf[idx]);
        } else {                             // residual out
          OutF[idx] = xres[idx] + rs * v;
        }
      }
    }
  }
}

// ---------------- chunked linear scan -------------------------------------
// (f,s)_t = [[l,0],[a*l,1-a]] (f,s)_{t-1} + [b, a*b]
__global__ __launch_bounds__(256) void scan_chunk_summary(
    const float* __restrict__ lam, const float* __restrict__ bg,
    const float* __restrict__ al,  const float* __restrict__ f_prev,
    float* __restrict__ Sp, float* __restrict__ Sq, float* __restrict__ Sr,
    float* __restrict__ Su, float* __restrict__ Sv)
{
  const int g = blockIdx.x * 256 + threadIdx.x;   // (b*CHUNKS + c)*DDIM + d
  const int d = g & (DDIM - 1);
  const int c = (g >> 10) & (CHUNKS - 1);
  const int b = g >> 16;
  float p = 1.f, q = 0.f, rr = 1.f, u = 0.f, v = 0.f;
  const float fp0 = (c == 0) ? f_prev[b * DDIM + d] : 0.0f;
  const size_t base = ((size_t)b * LSEQ + (size_t)c * CLEN) * DDIM + d;
  for (int i = 0; i < CLEN; ++i) {
    const size_t idx = base + (size_t)i * DDIM;
    float l = lam[idx], bb = bg[idx], a = al[idx];
    if (c == 0 && i == 0) bb += l * fp0;
    float np = l * p;
    float nu = l * u + bb;
    float nq = a * l * p + (1.f - a) * q;
    float nr = (1.f - a) * rr;
    float nv = a * nu + (1.f - a) * v;
    p = np; u = nu; q = nq; rr = nr; v = nv;
  }
  Sp[g] = p; Sq[g] = q; Sr[g] = rr; Su[g] = u; Sv[g] = v;
}

__global__ __launch_bounds__(256) void scan_combine(
    const float* __restrict__ Sp, const float* __restrict__ Sq,
    const float* __restrict__ Sr, const float* __restrict__ Su,
    const float* __restrict__ Sv, const float* __restrict__ s_prev,
    float* __restrict__ Cf, float* __restrict__ Cs)
{
  const int ch = blockIdx.x * 256 + threadIdx.x;  // b*DDIM + d
  const int b = ch >> 10, d = ch & (DDIM - 1);
  float f = 0.0f, s = s_prev[ch];
  for (int c = 0; c < CHUNKS; ++c) {
    const int g = (b * CHUNKS + c) * DDIM + d;
    Cf[g] = f; Cs[g] = s;
    float nf = Sp[g] * f + Su[g];
    float ns = Sq[g] * f + Sr[g] * s + Sv[g];
    f = nf; s = ns;
  }
}

__global__ __launch_bounds__(256) void scan_replay(
    const float* __restrict__ lam, const float* __restrict__ bg,
    const float* __restrict__ al,  const float* __restrict__ f_prev,
    const float* __restrict__ Cf,  const float* __restrict__ Cs,
    float* __restrict__ fst, float* __restrict__ sst,
    __bf16* __restrict__ fbf, __bf16* __restrict__ sbf,
    float* __restrict__ f_last, float* __restrict__ s_last)
{
  const int g = blockIdx.x * 256 + threadIdx.x;
  const int d = g & (DDIM - 1);
  const int c = (g >> 10) & (CHUNKS - 1);
  const int b = g >> 16;
  float f = Cf[g], s = Cs[g];
  const float fp0 = (c == 0) ? f_prev[b * DDIM + d] : 0.0f;
  const size_t base = ((size_t)b * LSEQ + (size_t)c * CLEN) * DDIM + d;
  for (int i = 0; i < CLEN; ++i) {
    const size_t idx = base + (size_t)i * DDIM;
    float l = lam[idx], bb = bg[idx], a = al[idx];
    if (c == 0 && i == 0) bb += l * fp0;
    f = l * f + bb;
    fst[idx] = f; fbf[idx] = (__bf16)f;
    s = s + a * (f - s);
    sst[idx] = s; sbf[idx] = (__bf16)s;
  }
  if (c == CHUNKS - 1) { f_last[b * DDIM + d] = f; s_last[b * DDIM + d] = s; }
}

// ---------------- launcher --------------------------------------------------
extern "C" void kernel_launch(void* const* d_in, const int* in_sizes, int n_in,
                              void* d_out, int out_size, void* d_ws, size_t ws_size,
                              hipStream_t stream) {
  const float* x         = (const float*)d_in[0];
  const float* f_prev    = (const float*)d_in[1];
  const float* s_prev    = (const float*)d_in[2];
  const float* w_lambda  = (const float*)d_in[3];
  const float* P_lambda  = (const float*)d_in[4];
  const float* W_beta    = (const float*)d_in[5];
  const float* W_alpha   = (const float*)d_in[6];
  const float* W_alpha_b = (const float*)d_in[7];
  const float* W_rho     = (const float*)d_in[8];
  const float* W_rho_b   = (const float*)d_in[9];
  const float* out_proj  = (const float*)d_in[10];
  const float* res_scale = (const float*)d_in[11];

  float* out    = (float*)d_out;
  float* f_last = out + (size_t)MTOK * DDIM;
  float* s_last = f_last + (size_t)BQ * DDIM;

  float* ws = (float*)d_ws;
  size_t off = 0;
  auto alloc = [&](size_t nf) { float* p = ws + off; off += nf; return p; };
  const size_t P = (size_t)MTOK * DDIM;          // 8M floats per plane
  const size_t G = (size_t)BQ * CHUNKS * DDIM;   // 262144
  float* lam = alloc(P);
  float* Bg  = alloc(P);
  float* alp = alloc(P);
  float* fst = alloc(P);
  float* sst = alloc(P);
  float *Sp = alloc(G), *Sq = alloc(G), *Sr = alloc(G);
  float *Su = alloc(G), *Sv = alloc(G), *Cf = alloc(G), *Cs = alloc(G);
  __bf16* xbf = (__bf16*)alloc(P / 2);
  __bf16* fbf = (__bf16*)alloc(P / 2);
  __bf16* sbf = (__bf16*)alloc(P / 2);
  const size_t WSZ = (size_t)DDIM * DDIM;
  __bf16* Plb = (__bf16*)alloc(WSZ / 2);
  __bf16* Wbb = (__bf16*)alloc(WSZ / 2);
  __bf16* Wab = (__bf16*)alloc(WSZ / 2);
  __bf16* Opb = (__bf16*)alloc(WSZ / 2);
  __bf16* Wrb = (__bf16*)alloc(3 * WSZ / 2);
  __bf16* hbf = (__bf16*)Bg;                     // Bg dead after replay

  // --- pre-convert operands to bf16 ---
  cvt_f32_bf16<<<(unsigned)(P / 2048), 256, 0, stream>>>(x, xbf);
  cvt_f32_bf16<<<(unsigned)(WSZ / 2048), 256, 0, stream>>>(P_lambda, Plb);
  cvt_f32_bf16<<<(unsigned)(WSZ / 2048), 256, 0, stream>>>(W_beta, Wbb);
  cvt_f32_bf16<<<(unsigned)(WSZ / 2048), 256, 0, stream>>>(W_alpha, Wab);
  cvt_f32_bf16<<<(unsigned)(WSZ / 2048), 256, 0, stream>>>(out_proj, Opb);
  cvt_f32_bf16<<<(unsigned)(3 * WSZ / 2048), 256, 0, stream>>>(W_rho, Wrb);

  dim3 gg(MTOK / 64, DDIM / 128);
  wmma_gemm_epi<0,false><<<gg,256,0,stream>>>(xbf,nullptr,nullptr,nullptr,Plb,
      w_lambda,nullptr,nullptr,nullptr,nullptr, lam, nullptr, DDIM);
  wmma_gemm_epi<1,false><<<gg,256,0,stream>>>(xbf,nullptr,nullptr,nullptr,Wbb,
      nullptr,nullptr,nullptr,nullptr,nullptr, Bg, nullptr, DDIM);
  wmma_gemm_epi<2,false><<<gg,256,0,stream>>>(xbf,nullptr,nullptr,nullptr,Wab,
      W_alpha_b,nullptr,nullptr,nullptr,nullptr, alp, nullptr, DDIM);

  scan_chunk_summary<<<(unsigned)(G / 256),256,0,stream>>>(lam,Bg,alp,f_prev,Sp,Sq,Sr,Su,Sv);
  scan_combine<<<(BQ*DDIM)/256,256,0,stream>>>(Sp,Sq,Sr,Su,Sv,s_prev,Cf,Cs);
  scan_replay<<<(unsigned)(G / 256),256,0,stream>>>(lam,Bg,alp,f_prev,Cf,Cs,
      fst,sst,fbf,sbf,f_last,s_last);

  wmma_gemm_epi<3,true><<<gg,256,0,stream>>>(nullptr,fbf,sbf,xbf,Wrb,
      W_rho_b,fst,sst,nullptr,nullptr, nullptr, hbf, 3 * DDIM);
  wmma_gemm_epi<4,false><<<gg,256,0,stream>>>(hbf,nullptr,nullptr,nullptr,Opb,
      nullptr,nullptr,nullptr,x,res_scale, out, nullptr, DDIM);
}
